// MultiHeadAttentionBlock_21388937134769
// MI455X (gfx1250) — matrix operand order repaired and checked by reading, not compiled, not run on hardware
//
#include <hip/hip_runtime.h>
#include <stdint.h>

#define DEV __device__ __forceinline__

typedef __attribute__((ext_vector_type(16))) _Float16 half16;
typedef __attribute__((ext_vector_type(8)))  _Float16 half8;
typedef __attribute__((ext_vector_type(8)))  float    float8;
typedef int v4i __attribute__((__vector_size__(16)));

constexpr int Bc = 8;
constexpr int Sc = 1024;
constexpr int Dc = 256;
constexpr int Hc = 8;
constexpr int HD = Hc * Dc;          // 2048
constexpr int Mtot = Bc * Sc;        // 8192

// ---------------- async copy helpers (gfx1250 GLOBAL_LOAD_ASYNC_TO_LDS) -----------

#if __has_builtin(__builtin_amdgcn_global_load_async_to_lds_b128) && \
    __has_builtin(__builtin_amdgcn_s_wait_asynccnt)
#define USE_ASYNC 1
#else
#define USE_ASYNC 0
#endif

DEV void async_b128(const _Float16* g, _Float16* l) {
#if USE_ASYNC
    __builtin_amdgcn_global_load_async_to_lds_b128(
        (__attribute__((address_space(1))) v4i*)(uintptr_t)g,
        (__attribute__((address_space(3))) v4i*)(uint32_t)(uintptr_t)l,
        0, 0);
#else
    *(half8*)l = *(const half8*)g;
#endif
}

DEV void wait_async_all() {
#if USE_ASYNC
    __builtin_amdgcn_s_wait_asynccnt(0);
#endif
}

// ---------------- fragment helpers (per CDNA5 ISA layouts) --------------

// A matrix 16x32 (MxK), f16, row-major source with leading dim `ld` (elements).
// lane m = lane&15 ; K-base = 0 (lanes 0-15) or 8 (lanes 16-31)
// VGPR 0..3 : K = kb+0..kb+7   ; VGPR 4..7 : K = kb+16..kb+23
DEV half16 load_a_frag(const _Float16* base, int ld, int lane) {
    int m  = lane & 15;
    int kb = (lane < 16) ? 0 : 8;
    const _Float16* p = base + (size_t)m * ld + kb;
    half8 lo = *(const half8*)(p);
    half8 hi = *(const half8*)(p + 16);
    half16 a;
#pragma unroll
    for (int i = 0; i < 8; ++i) { a[i] = lo[i]; a[8 + i] = hi[i]; }
    return a;
}

// B matrix 32x16 (KxN), f16, loaded from a row-major matrix whose ROW index is N
// and COLUMN index is K (i.e. B = source^T). lane n = lane&15 ;
// K-base = 0 (lanes 0-15) or 8 (lanes 16-31); K = kb+0..7, kb+16..23 (same as A).
DEV half16 load_b_frag(const _Float16* base, int ld, int lane) {
    int n  = lane & 15;
    int kb = (lane < 16) ? 0 : 8;
    const _Float16* p = base + (size_t)n * ld + kb;
    half8 lo = *(const half8*)(p);
    half8 hi = *(const half8*)(p + 16);
    half16 b;
#pragma unroll
    for (int i = 0; i < 8; ++i) { b[i] = lo[i]; b[8 + i] = hi[i]; }
    return b;
}

DEV float8 f8_zero() {
    float8 z;
#pragma unroll
    for (int i = 0; i < 8; ++i) z[i] = 0.0f;
    return z;
}

DEV float8 wmma16(half16 a, half16 b, float8 c) {
    return __builtin_amdgcn_wmma_f32_16x16x32_f16(false, a, false, b, (short)0, c,
                                                  false, false);
}

// reductions across the 16-lane half-groups (C tile: lanes 0-15 = cols, rows 0-7;
// lanes 16-31 = cols, rows 8-15). xor masks 1,2,4,8 stay inside each half.
DEV float rowmax16(float v) {
    v = fmaxf(v, __shfl_xor(v, 1, 32));
    v = fmaxf(v, __shfl_xor(v, 2, 32));
    v = fmaxf(v, __shfl_xor(v, 4, 32));
    v = fmaxf(v, __shfl_xor(v, 8, 32));
    return v;
}
DEV float rowsum16(float v) {
    v += __shfl_xor(v, 1, 32);
    v += __shfl_xor(v, 2, 32);
    v += __shfl_xor(v, 4, 32);
    v += __shfl_xor(v, 8, 32);
    return v;
}

// ---------------- kernel 1 : f32 -> f16 conversion --------------------------------

__global__ void k_convert(const float* __restrict__ x,  const float* __restrict__ wq,
                          const float* __restrict__ wk, const float* __restrict__ wv,
                          const float* __restrict__ wp,
                          _Float16* xb, _Float16* wqb, _Float16* wkb,
                          _Float16* wvb, _Float16* wpb) {
    int gid = blockIdx.x * blockDim.x + threadIdx.x;
    const int NX = Mtot * Dc;        // 2097152
    const int NW = Hc * Dc * Dc;     // 524288
    const int NP = Dc * HD;          // 524288
    if (gid < NX) xb[gid] = (_Float16)x[gid];
    if (gid < NW) {
        wqb[gid] = (_Float16)wq[gid];
        wkb[gid] = (_Float16)wk[gid];
        wvb[gid] = (_Float16)wv[gid];
    }
    if (gid < NP) wpb[gid] = (_Float16)wp[gid];
}

// ---------------- kernel 2 : QKV projection GEMMs ---------------------------------
// grid (128, 4, 24) : x = 64-row block, y = 64-col strip, z = mat*8 + head
// Weight 64x32 B-tiles async double-buffered in LDS (shared by 4 waves);
// A fragment prefetched into registers one k-step ahead.
// Q is pre-scaled by 1/sqrt(D); V stored transposed.

__global__ __launch_bounds__(128) void k_qkv(
    const _Float16* __restrict__ xb,
    const _Float16* __restrict__ wqb, const _Float16* __restrict__ wkb,
    const _Float16* __restrict__ wvb,
    const float* __restrict__ bq, const float* __restrict__ bk,
    const float* __restrict__ bvp,
    _Float16* Qb, _Float16* Kb, _Float16* Vt) {
    __shared__ __align__(32) _Float16 wtile[2][64 * 32];   // 2 x 4 KiB
    int tid = threadIdx.x;
    int lane = tid & 31, wid = tid >> 5;
    int r0 = blockIdx.x * 64 + wid * 16;   // global row in [0, 8192)
    int n0 = blockIdx.y * 64;              // output-col strip base
    int mh = blockIdx.z;
    int mat = mh >> 3, h = mh & 7;

    const _Float16* W    = (mat == 0) ? wqb : (mat == 1) ? wkb : wvb;
    const float*    bias = (mat == 0) ? bq  : (mat == 1) ? bk  : bvp;
    const _Float16* wbase = W + (size_t)h * Dc * Dc;

    // prefetch first W tile (rows n0..n0+63, K = 0..31)
    for (int i = tid; i < 64 * 4; i += 128) {
        int rr = i >> 2, cc = (i & 3) * 8;
        async_b128(wbase + (size_t)(n0 + rr) * Dc + cc, &wtile[0][rr * 32 + cc]);
    }

    int n = lane & 15;
    float8 acc[4];
#pragma unroll
    for (int j = 0; j < 4; ++j) {
        float bval = bias[h * Dc + n0 + j * 16 + n];
#pragma unroll
        for (int v = 0; v < 8; ++v) acc[j][v] = bval;
    }

    const _Float16* abase = xb + (size_t)r0 * Dc;
    half16 a = load_a_frag(abase, Dc, lane);
#pragma unroll
    for (int kk = 0; kk < 8; ++kk) {
        int cur = kk & 1;
        wait_async_all();
        __syncthreads();
        if (kk + 1 < 8) {
            for (int i = tid; i < 64 * 4; i += 128) {
                int rr = i >> 2, cc = (i & 3) * 8;
                async_b128(wbase + (size_t)(n0 + rr) * Dc + (kk + 1) * 32 + cc,
                           &wtile[cur ^ 1][rr * 32 + cc]);
            }
        }
        half16 anext = a;
        if (kk + 1 < 8) anext = load_a_frag(abase + (kk + 1) * 32, Dc, lane);
#pragma unroll
        for (int j = 0; j < 4; ++j) {
            half16 b = load_b_frag(&wtile[cur][j * 16 * 32], 32, lane);
            acc[j] = wmma16(a, b, acc[j]);
        }
        a = anext;
        __syncthreads();
    }
    if (mat == 0) {
#pragma unroll
        for (int j = 0; j < 4; ++j)
#pragma unroll
            for (int v = 0; v < 8; ++v) acc[j][v] *= 0.0625f;   // 1/sqrt(256)
    }

    int bb = r0 / Sc, s = r0 % Sc;
    int mb = (lane < 16) ? 0 : 8;
    if (mat < 2) {
        _Float16* out = ((mat == 0) ? Qb : Kb) +
                        (((size_t)(bb * Hc + h)) * Sc + s) * Dc;
#pragma unroll
        for (int j = 0; j < 4; ++j)
#pragma unroll
            for (int v = 0; v < 8; ++v)
                out[(size_t)(mb + v) * Dc + n0 + j * 16 + n] = (_Float16)acc[j][v];
    } else {
        // V transposed: Vt[(bh)*D*S + e*S + t]
        _Float16* out = Vt + ((size_t)(bb * Hc + h)) * Dc * Sc;
#pragma unroll
        for (int j = 0; j < 4; ++j)
#pragma unroll
            for (int v = 0; v < 8; ++v)
                out[(size_t)(n0 + j * 16 + n) * Sc + s + mb + v] = (_Float16)acc[j][v];
    }
}

// ---------------- kernel 3 : flash attention -------------------------------------
// grid (16, 64) : x = 64-query-row block, y = b*H + h. 4 waves, 16 rows each.
// K / V^T tiles double-buffered in LDS via async copies (ASYNCcnt).

__global__ __launch_bounds__(128) void k_attn(
    const _Float16* __restrict__ Qb, const _Float16* __restrict__ Kb,
    const _Float16* __restrict__ Vt, _Float16* Ocat) {
    __shared__ __align__(32) _Float16 qs[64 * Dc];       // 32 KiB
    __shared__ __align__(32) _Float16 ks[2][32 * Dc];    // 2 x 16 KiB
    __shared__ __align__(32) _Float16 vs[2][Dc * 32];    // 2 x 16 KiB
    __shared__ __align__(32) _Float16 ps[4][16 * 32];    // 4 KiB (per-wave P tile)

    int tid = threadIdx.x;
    int lane = tid & 31, wid = tid >> 5;
    int s0 = blockIdx.x * 64;
    int bh = blockIdx.y;
    int bb = bh >> 3, h = bh & 7;

    const _Float16* qsrc = Qb + (((size_t)bh) * Sc + s0) * Dc;
    const _Float16* kb0  = Kb + ((size_t)bh) * Sc * Dc;
    const _Float16* vb0  = Vt + ((size_t)bh) * Dc * Sc;

    // async stage: Q block + first K / V^T tiles
    for (int i = tid; i < 64 * Dc / 8; i += 128) async_b128(qsrc + i * 8, qs + i * 8);
    for (int i = tid; i < 32 * Dc / 8; i += 128) async_b128(kb0 + i * 8, &ks[0][i * 8]);
    for (int i = tid; i < Dc * 4; i += 128) {
        int e = i >> 2, c = (i & 3) * 8;
        async_b128(vb0 + (size_t)e * Sc + c, &vs[0][e * 32 + c]);
    }

    int m0 = wid * 16;
    int n = lane & 15;
    int mb = (lane < 16) ? 0 : 8;

    float mrun[8], lrun[8];
#pragma unroll
    for (int v = 0; v < 8; ++v) { mrun[v] = -1e30f; lrun[v] = 0.0f; }
    float8 oacc[16];
#pragma unroll
    for (int et = 0; et < 16; ++et) oacc[et] = f8_zero();

    _Float16* pw = &ps[wid][0];

    for (int it = 0; it < Sc / 32; ++it) {
        int cur = it & 1;
        wait_async_all();
        __syncthreads();                       // tile[cur] ready for everyone

        // prefetch next K / V^T tiles into the other buffer
        if (it + 1 < Sc / 32) {
            int nxt = cur ^ 1;
            int t1 = (it + 1) * 32;
            for (int i = tid; i < 32 * Dc / 8; i += 128)
                async_b128(kb0 + (size_t)t1 * Dc + i * 8, &ks[nxt][i * 8]);
            for (int i = tid; i < Dc * 4; i += 128) {
                int e = i >> 2, c = (i & 3) * 8;
                async_b128(vb0 + (size_t)e * Sc + t1 + c, &vs[nxt][e * 32 + c]);
            }
        }

        // ---- scores for 32 keys: two 16x16 C tiles (operands from LDS)
        float8 sc0 = f8_zero(), sc1 = f8_zero();
#pragma unroll
        for (int kk = 0; kk < 8; ++kk) {
            half16 a  = load_a_frag(&qs[(size_t)m0 * Dc + kk * 32], Dc, lane);
            half16 b0 = load_b_frag(&ks[cur][kk * 32], Dc, lane);
            half16 b1 = load_b_frag(&ks[cur][16 * Dc + kk * 32], Dc, lane);
            sc0 = wmma16(a, b0, sc0);
            sc1 = wmma16(a, b1, sc1);
        }

        // ---- online softmax update (per row v within each 16-lane group)
        float alpha[8];
#pragma unroll
        for (int v = 0; v < 8; ++v) {
            float rm   = rowmax16(fmaxf(sc0[v], sc1[v]));
            float mnew = fmaxf(mrun[v], rm);
            alpha[v]   = __expf(mrun[v] - mnew);
            mrun[v]    = mnew;
            float p0 = __expf(sc0[v] - mnew);
            float p1 = __expf(sc1[v] - mnew);
            sc0[v] = p0; sc1[v] = p1;
            lrun[v] = lrun[v] * alpha[v] + rowsum16(p0 + p1);
        }
#pragma unroll
        for (int et = 0; et < 16; ++et)
#pragma unroll
            for (int v = 0; v < 8; ++v) oacc[et][v] *= alpha[v];

        // ---- re-swizzle P (C layout) into A-fragment layout via LDS
#pragma unroll
        for (int v = 0; v < 8; ++v) {
            pw[(mb + v) * 32 + n]      = (_Float16)sc0[v];
            pw[(mb + v) * 32 + 16 + n] = (_Float16)sc1[v];
        }
        __syncthreads();
        half16 pa = load_a_frag(pw, 32, lane);

        // ---- O += P @ V  (B fragments from the staged V^T tile, ld = 32)
#pragma unroll
        for (int et = 0; et < 16; ++et) {
            half16 bvf = load_b_frag(&vs[cur][et * 16 * 32], 32, lane);
            oacc[et] = wmma16(pa, bvf, oacc[et]);
        }
        __syncthreads();                       // done reading tile[cur]
    }

    float linv[8];
#pragma unroll
    for (int v = 0; v < 8; ++v) linv[v] = 1.0f / lrun[v];

    // store concatenated heads: Ocat[b, s, h*D + e]
    _Float16* obase = Ocat + ((size_t)bb * Sc + s0 + m0) * HD + h * Dc;
#pragma unroll
    for (int et = 0; et < 16; ++et)
#pragma unroll
        for (int v = 0; v < 8; ++v)
            obase[(size_t)(mb + v) * HD + et * 16 + n] =
                (_Float16)(oacc[et][v] * linv[v]);
}

// ---------------- kernel 4 : output projection -----------------------------------
// out[r, d] = sum_c Ocat[r, c] * Wp[d, c]  ; grid (128, 4), wave = 16x64 strip.
// Wp 64x32 B-tiles double-buffered in LDS via async copies (shared by 4 waves);
// A fragment prefetched into registers one k-step ahead.

__global__ __launch_bounds__(128) void k_proj(
    const _Float16* __restrict__ Ocat, const _Float16* __restrict__ wpb,
    float* __restrict__ out) {
    __shared__ __align__(32) _Float16 bs[2][64 * 32];    // 2 x 4 KiB
    int tid = threadIdx.x;
    int lane = tid & 31, wid = tid >> 5;
    int r0 = blockIdx.x * 64 + wid * 16;
    int n0 = blockIdx.y * 64;

    // prefetch first B tile (rows n0..n0+63 of Wp, K = 0..31)
    for (int i = tid; i < 64 * 4; i += 128) {
        int rr = i >> 2, cc = (i & 3) * 8;
        async_b128(wpb + (size_t)(n0 + rr) * HD + cc, &bs[0][rr * 32 + cc]);
    }

    float8 acc[4];
#pragma unroll
    for (int j = 0; j < 4; ++j) acc[j] = f8_zero();

    const _Float16* abase = Ocat + (size_t)r0 * HD;
    half16 a = load_a_frag(abase, HD, lane);
    for (int kk = 0; kk < HD / 32; ++kk) {
        int cur = kk & 1;
        wait_async_all();
        __syncthreads();
        if (kk + 1 < HD / 32) {
            for (int i = tid; i < 64 * 4; i += 128) {
                int rr = i >> 2, cc = (i & 3) * 8;
                async_b128(wpb + (size_t)(n0 + rr) * HD + (kk + 1) * 32 + cc,
                           &bs[cur ^ 1][rr * 32 + cc]);
            }
        }
        half16 anext = a;
        if (kk + 1 < HD / 32) anext = load_a_frag(abase + (size_t)(kk + 1) * 32, HD, lane);
#pragma unroll
        for (int j = 0; j < 4; ++j) {
            half16 b = load_b_frag(&bs[cur][j * 16 * 32], 32, lane);
            acc[j] = wmma16(a, b, acc[j]);
        }
        a = anext;
        __syncthreads();
    }

    int n = lane & 15;
    int mb = (lane < 16) ? 0 : 8;
#pragma unroll
    for (int j = 0; j < 4; ++j)
#pragma unroll
        for (int v = 0; v < 8; ++v)
            out[(size_t)(r0 + mb + v) * Dc + n0 + j * 16 + n] = acc[j][v];
}

// ---------------- host launcher ---------------------------------------------------

extern "C" void kernel_launch(void* const* d_in, const int* in_sizes, int n_in,
                              void* d_out, int out_size, void* d_ws, size_t ws_size,
                              hipStream_t stream) {
    const float* x  = (const float*)d_in[0];
    const float* Wq = (const float*)d_in[1];
    const float* Wk = (const float*)d_in[2];
    const float* Wv = (const float*)d_in[3];
    const float* bq = (const float*)d_in[4];
    const float* bk = (const float*)d_in[5];
    const float* bv = (const float*)d_in[6];
    const float* Wp = (const float*)d_in[7];
    float* out = (float*)d_out;

    char* ws = (char*)d_ws;
    _Float16* xb  = (_Float16*)(ws + (0ull   << 20));  //  4 MiB  x f16
    _Float16* wqb = (_Float16*)(ws + (4ull   << 20));  //  1 MiB
    _Float16* wkb = (_Float16*)(ws + (5ull   << 20));  //  1 MiB
    _Float16* wvb = (_Float16*)(ws + (6ull   << 20));  //  1 MiB
    _Float16* wpb = (_Float16*)(ws + (7ull   << 20));  //  1 MiB
    _Float16* Qb  = (_Float16*)(ws + (8ull   << 20));  // 32 MiB  [B,H,S,D]
    _Float16* Kb  = (_Float16*)(ws + (40ull  << 20));  // 32 MiB  [B,H,S,D]
    _Float16* Vt  = (_Float16*)(ws + (72ull  << 20));  // 32 MiB  [B,H,D,S]
    _Float16* Oc  = (_Float16*)(ws + (104ull << 20));  // 32 MiB  [B,S,H*D]

    k_convert<<<8192, 256, 0, stream>>>(x, Wq, Wk, Wv, Wp, xb, wqb, wkb, wvb, wpb);
    k_qkv<<<dim3(128, 4, 24), 128, 0, stream>>>(xb, wqb, wkb, wvb, bq, bk, bv,
                                                Qb, Kb, Vt);
    k_attn<<<dim3(16, 64), 128, 0, stream>>>(Qb, Kb, Vt, Oc);
    k_proj<<<dim3(128, 4), 128, 0, stream>>>(Oc, wpb, out);
}
